// WaveBandNormB_33638183862718
// MI455X (gfx1250) — compile-verified, matrix operand order checked by reading, not run on hardware
//
#include <hip/hip_runtime.h>
#include <hip/hip_bf16.h>
#include <math.h>

typedef __attribute__((ext_vector_type(16))) __bf16 v16bf;
typedef __attribute__((ext_vector_type(8)))  float  v8f;
typedef __attribute__((ext_vector_type(4)))  unsigned int u32x4;
typedef __attribute__((ext_vector_type(8)))  int i32x8;
typedef __attribute__((ext_vector_type(4)))  int i32x4;

#define B_   32
#define T_   720
#define C_   512
#define PH   30
#define PF   14
#define PL   24
#define HID  256
#define NIN  7
#define NOUT 4
#define SIGMIN 1e-3f
#define EPSV   1e-6f
#define TC   8          // channels per workgroup in SWT kernel
#define MT   32         // rows per workgroup in GRU kernel

#if defined(__has_builtin)
#if __has_builtin(__builtin_amdgcn_tensor_load_to_lds) && __has_builtin(__builtin_amdgcn_s_wait_tensorcnt)
#define HAVE_TDM 1
#endif
#endif

// ---------------- f32 -> bf16 weight conversion ----------------
__global__ void conv_bf16_kernel(const float* __restrict__ encW,
                                 const float* __restrict__ decW,
                                 unsigned short* __restrict__ encO,
                                 unsigned short* __restrict__ decO, int n) {
  int i = blockIdx.x * blockDim.x + threadIdx.x;
  __bf16* eo  = reinterpret_cast<__bf16*>(encO);
  __bf16* dfo = reinterpret_cast<__bf16*>(decO);
  if (i < n) {
    eo[i]  = (__bf16)encW[i];
    dfo[i] = (__bf16)decW[i];
  }
}

// ---------------- SWT + patch stats + x_norm + GRU features ----------------
__global__ __launch_bounds__(256) void swt_stats_kernel(
    const float* __restrict__ x, float* __restrict__ xnorm,
    float* __restrict__ feat, float* __restrict__ last) {
  __shared__ __align__(16) float sA[TC * T_], sB[TC * T_], sXN[TC * T_];
  __shared__ float sMuL[TC * PH], sSigL[TC * PH], sSig1[TC * PH], sSig2[TC * PH],
                   sEH[TC * PH], sEt[TC * PH];
  __shared__ float sMug[TC], sSigg[TC];

  const int tid = threadIdx.x;
  const int b   = blockIdx.x / (C_ / TC);
  const int c0  = (blockIdx.x % (C_ / TC)) * TC;

  // vectorized load: 2 float4 per t cover 8 channels
  for (int i = tid; i < 2 * T_; i += 256) {
    int q = i & 1, t = i >> 1;
    float4 v = *(const float4*)&x[((size_t)b * T_ + t) * C_ + c0 + q * 4];
    sA[(q * 4 + 0) * T_ + t] = v.x;
    sA[(q * 4 + 1) * T_ + t] = v.y;
    sA[(q * 4 + 2) * T_ + t] = v.z;
    sA[(q * 4 + 3) * T_ + t] = v.w;
  }
  for (int i = tid; i < TC * T_; i += 256) sXN[i] = 0.f;
  __syncthreads();

  float* ap = sA;
  float* an = sB;
  for (int j = 1; j <= 3; ++j) {
    const int step = 1 << (j - 1);
    const int pad  = (step + 1) >> 1;
    for (int i = tid; i < TC * T_; i += 256) {
      int c = i % TC, t = i / TC;
      int k0 = t - pad;        k0 = k0 < 0 ? -k0 : (k0 >= T_ ? 2 * T_ - 2 - k0 : k0);
      int k1 = t - pad + step; k1 = k1 < 0 ? -k1 : (k1 >= T_ ? 2 * T_ - 2 - k1 : k1);
      an[c * T_ + t] = 0.5f * (ap[c * T_ + k0] + ap[c * T_ + k1]);
    }
    __syncthreads();
    if (tid < TC * PH) {
      int c = tid % TC, p = tid / TC;
      float s = 0.f, ss = 0.f;
      for (int q = 0; q < PL; ++q) {
        float d = ap[c * T_ + p * PL + q] - an[c * T_ + p * PL + q];
        s += d; ss += d * d;
      }
      float mu  = s / PL;
      float var = (ss - PL * mu * mu) / (PL - 1);
      float sg  = fmaxf(sqrtf(fmaxf(var, 0.f)), SIGMIN);
      float E   = ss / PL;
      if (j == 1)      { sEH[c * PH + p] = E; sEt[c * PH + p] = E; }
      else             { sEt[c * PH + p] += E;
                         if (j == 2) sSig1[c * PH + p] = sg;
                         else        sSig2[c * PH + p] = sg; }
    }
    __syncthreads();
    for (int i = tid; i < TC * T_; i += 256) {
      int c = i % TC, t = i / TC;
      float d = ap[c * T_ + t] - an[c * T_ + t];
      if (j == 1) sXN[c * T_ + t] += d;
      else {
        int p = t / PL;
        float sg = (j == 2) ? sSig1[c * PH + p] : sSig2[c * PH + p];
        sXN[c * T_ + t] += d / sg;
      }
    }
    __syncthreads();
    float* tmp = ap; ap = an; an = tmp;
  }

  // L = ap : patch + global stats
  if (tid < TC * PH) {
    int c = tid % TC, p = tid / TC;
    float s = 0.f, ss = 0.f;
    for (int q = 0; q < PL; ++q) { float v = ap[c * T_ + p * PL + q]; s += v; ss += v * v; }
    float mu  = s / PL;
    float var = (ss - PL * mu * mu) / (PL - 1);
    float sg  = fmaxf(sqrtf(fmaxf(var, 0.f)), SIGMIN);
    sMuL[c * PH + p]  = mu;
    sSigL[c * PH + p] = sg;
    sEt[c * PH + p]  += ss / PL;
  }
  if (tid < TC) {
    int c = tid;
    float s = 0.f, ss = 0.f;
    for (int t = 0; t < T_; ++t) { float v = ap[c * T_ + t]; s += v; ss += v * v; }
    float mu  = s / T_;
    float var = (ss - T_ * mu * mu) / (T_ - 1);
    sMug[c]  = mu;
    sSigg[c] = fmaxf(sqrtf(fmaxf(var, 0.f)), SIGMIN);
  }
  __syncthreads();

  for (int i = tid; i < 2 * T_; i += 256) {   // vectorized store
    int q = i & 1, t = i >> 1, p = t / PL;
    float4 v;
    float* vv = &v.x;
    #pragma unroll
    for (int k = 0; k < 4; ++k) {
      int c = q * 4 + k;
      vv[k] = sXN[c * T_ + t] + (ap[c * T_ + t] - sMuL[c * PH + p]) / sSigL[c * PH + p];
    }
    *(float4*)&xnorm[((size_t)b * T_ + t) * C_ + c0 + q * 4] = v;
  }
  if (tid < TC * PH) {
    int c = tid % TC, p = tid / TC;
    int row = b * C_ + c0 + c;
    float* f = feat + ((size_t)row * PH + p) * NIN;
    f[0] = sMuL[c * PH + p];
    f[1] = logf(sSigL[c * PH + p]);
    f[2] = logf(sSig1[c * PH + p]);
    f[3] = logf(sSig2[c * PH + p]);
    f[4] = sEH[c * PH + p] / (sEt[c * PH + p] + EPSV);
    f[5] = sMug[c];
    f[6] = logf(sSigg[c]);
  }
  if (tid < TC) {
    int c = tid, row = b * C_ + c0 + c;
    last[row * 4 + 0] = sMuL[c * PH + PH - 1];
    last[row * 4 + 1] = logf(sSigL[c * PH + PH - 1]);
    last[row * 4 + 2] = logf(sSig1[c * PH + PH - 1]);
    last[row * 4 + 3] = logf(sSig2[c * PH + PH - 1]);
  }
}

// two contiguous 16B runs -> one 16x32 bf16 fragment (K pairs per VGPR)
__device__ __forceinline__ v16bf load_frag16(const __bf16* p0, const __bf16* p1) {
  struct U { uint4 a, b; } u;
  u.a = *(const uint4*)p0;
  u.b = *(const uint4*)p1;
  return __builtin_bit_cast(v16bf, u);
}

#ifdef HAVE_TDM
// TDM: 1-D tile of 16384 2-byte elements (one 64x256 bf16 weight chunk) -> LDS
__device__ __forceinline__ void tdm_load_chunk(const __bf16* gsrc, unsigned lds_off) {
  unsigned long long ga = (unsigned long long)gsrc;
  u32x4 g0;
  g0[0] = 1u;                                    // count=1, user mode
  g0[1] = lds_off;                               // lds_addr (bytes)
  g0[2] = (unsigned)(ga & 0xFFFFFFFFull);        // global_addr[31:0]
  g0[3] = (unsigned)((ga >> 32) & 0x1FFFFFFull)  // global_addr[56:32]
          | (2u << 30);                          // type=2 (image)
  i32x8 g1;
  g1[0] = 0x10000;           // data_size=1 (2 bytes/elem)
  g1[1] = (int)0x40000000;   // tensor_dim0[15:0]=0x4000 in bits[31:16]
  g1[2] = 0x10000;           // tensor_dim0 hi=0, tensor_dim1=1 in bits[31:16]
  g1[3] = (int)0x40000000;   // tile_dim0=0x4000 in bits[31:16]
  g1[4] = 1;                 // tile_dim1=1
  g1[5] = 16384;             // tensor_dim0_stride low32
  g1[6] = 0;
  g1[7] = 0;
  i32x4 z = {0, 0, 0, 0};
#if defined(__clang_major__) && (__clang_major__ >= 23)
  i32x8 z8 = {0, 0, 0, 0, 0, 0, 0, 0};
  __builtin_amdgcn_tensor_load_to_lds(g0, g1, z, z, z8, 0);
#else
  __builtin_amdgcn_tensor_load_to_lds(g0, g1, z, z, 0);
#endif
}
#endif

// ---------------- persistent GRU enc+dec with bf16 WMMA ----------------
__global__ __launch_bounds__(256) void gru_wmma_kernel(
    const float* __restrict__ feat, const float* __restrict__ last,
    const unsigned short* __restrict__ encWhhU, const unsigned short* __restrict__ decWhhU,
    const float* __restrict__ encWih, const float* __restrict__ encBih, const float* __restrict__ encBhh,
    const float* __restrict__ decWih, const float* __restrict__ decBih, const float* __restrict__ decBhh,
    const float* __restrict__ futEmb, const float* __restrict__ headW, const float* __restrict__ headB,
    float* __restrict__ pred) {
  __shared__ __align__(16) float  Hs[MT * HID];     // 32 KB  hidden state (f32)
  __shared__ __align__(16) __bf16 Hbf[MT * HID];    // 16 KB  bf16 mirror for WMMA A
  __shared__ __align__(16) __bf16 Wlds[64 * HID];   // 32 KB  64-gate weight chunk
  __shared__ __align__(16) float  Gbuf[3][MT * 64]; // 24 KB  r/z/n partial gates
  __shared__ float xrow[MT * 8];
  __shared__ float gid[3 * HID];

  const int tid  = threadIdx.x;
  const int lane = tid & 31;
  const int wave = tid >> 5;
  const int row0 = blockIdx.x * MT;
  const __bf16* encWhh = reinterpret_cast<const __bf16*>(encWhhU);
  const __bf16* decWhh = reinterpret_cast<const __bf16*>(decWhhU);
  const unsigned wldsOff = (unsigned)(unsigned long long)(void*)&Wlds[0];
  (void)wldsOff;

  for (int i = tid; i < MT * HID; i += 256) { Hs[i] = 0.f; Hbf[i] = (__bf16)0.f; }
  __syncthreads();

  const int m  = wave & 1;       // wave's M fragment (rows m*16 .. m*16+15)
  const int nf = wave >> 1;      // wave's N fragment within 64-gate chunk
  const int lm = lane & 15, half = lane >> 4;

  for (int t = 0; t < PH + PF; ++t) {
    const bool enc = (t < PH);
    if (enc) {
      if (tid < MT * NIN) {
        int r = tid / NIN, f = tid % NIN;
        xrow[r * 8 + f] = feat[((size_t)(row0 + r) * PH + t) * NIN + f];
      }
    } else {
      int td = t - PH;
      for (int g = tid; g < 3 * HID; g += 256) {
        float s = 0.f;
        #pragma unroll
        for (int f = 0; f < 16; ++f) s += futEmb[td * 16 + f] * decWih[g * 16 + f];
        gid[g] = s;
      }
    }
    __syncthreads();

    // A fragments (hidden-state tile) cached in registers for the whole step
    v16bf afrag[8];
    {
      const __bf16* ab = Hbf + (size_t)(m * 16 + lm) * HID;
      #pragma unroll
      for (int kk = 0; kk < 8; ++kk)
        afrag[kk] = load_frag16(ab + kk * 32 + half * 8, ab + kk * 32 + 16 + half * 8);
    }

    float hnew[4][8];
    const __bf16* Wbf = enc ? encWhh : decWhh;

    for (int hb = 0; hb < 4; ++hb) {             // 64 hidden units per block
      for (int gg = 0; gg < 3; ++gg) {           // r, z, n gate groups
        const __bf16* gsrc = Wbf + (size_t)(gg * HID + hb * 64) * HID;
        // ---- stage 32KB weight chunk into LDS ----
#ifdef HAVE_TDM
        if (wave == 0) {
          tdm_load_chunk(gsrc, wldsOff);
          __builtin_amdgcn_s_wait_tensorcnt(0);
        }
#else
        {
          const uint4* g4 = (const uint4*)gsrc;
          uint4* s4 = (uint4*)Wlds;
          for (int i = tid; i < (64 * HID) / 8; i += 256) s4[i] = g4[i];
        }
#endif
        // prefetch next chunk into cache
        {
          int nxt = (hb * 3 + gg + 1) % 12;
          const char* np = (const char*)(Wbf + (size_t)((nxt % 3) * HID + (nxt / 3) * 64) * HID);
          __builtin_prefetch(np + tid * 128, 0, 0);
        }
        __syncthreads();

        // ---- 16x16x32 bf16 WMMA, K=256 fully unrolled ----
        {
          const __bf16* bbase = Wlds + (size_t)(nf * 16 + lm) * HID;
          v8f acc = {};
          #pragma unroll
          for (int kk = 0; kk < 8; ++kk) {
            v16bf bb = load_frag16(bbase + kk * 32 + half * 8,
                                   bbase + kk * 32 + 16 + half * 8);
            acc = __builtin_amdgcn_wmma_f32_16x16x32_bf16(
                false, afrag[kk], false, bb, (short)0, acc, false, false);
          }
          const int col = lane & 15, rsel = lane >> 4;
          #pragma unroll
          for (int g = 0; g < 8; ++g)
            Gbuf[gg][(m * 16 + g + rsel * 8) * 64 + nf * 16 + col] = acc[g];
        }
        __syncthreads();
      }
      // ---- fused GRU update for hidden [hb*64, hb*64+64) ----
      for (int e = 0; e < 8; ++e) {
        int i  = tid + e * 256;
        int r  = i >> 6, hl = i & 63;
        int h  = hb * 64 + hl;
        float gir, giz, gin;
        if (enc) {
          gir = giz = gin = 0.f;
          #pragma unroll
          for (int f = 0; f < NIN; ++f) {
            float xv = xrow[r * 8 + f];
            gir += xv * encWih[(h)*NIN + f];
            giz += xv * encWih[(256 + h) * NIN + f];
            gin += xv * encWih[(512 + h) * NIN + f];
          }
        } else {
          gir = gid[h]; giz = gid[256 + h]; gin = gid[512 + h];
        }
        const float* bih = enc ? encBih : decBih;
        const float* bhh = enc ? encBhh : decBhh;
        float hr = Gbuf[0][r * 64 + hl] + bhh[h];
        float hz = Gbuf[1][r * 64 + hl] + bhh[256 + h];
        float hn = Gbuf[2][r * 64 + hl] + bhh[512 + h];
        float ir = gir + bih[h], iz = giz + bih[256 + h], nn = gin + bih[512 + h];
        float rg = 1.f / (1.f + expf(-(ir + hr)));
        float zg = 1.f / (1.f + expf(-(iz + hz)));
        float ng = tanhf(nn + rg * hn);
        hnew[hb][e] = (1.f - zg) * ng + zg * Hs[r * HID + h];
      }
    }
    __syncthreads();
    #pragma unroll
    for (int hb = 0; hb < 4; ++hb)
      for (int e = 0; e < 8; ++e) {
        int i = tid + e * 256;
        int r = i >> 6, h = hb * 64 + (i & 63);
        float v = hnew[hb][e];
        Hs[r * HID + h]  = v;
        Hbf[r * HID + h] = (__bf16)v;
      }
    __syncthreads();

    if (!enc) {  // head + pred write
      int td = t - PH;
      if (tid < MT * NOUT) {
        int r = tid >> 2, o = tid & 3;
        float s = headB[o];
        for (int h = 0; h < HID; ++h) s += Hs[r * HID + h] * headW[o * HID + h];
        int row = row0 + r, bb2 = row >> 9, cc = row & 511;
        pred[(((size_t)bb2 * PF + td) * NOUT + o) * C_ + cc] = last[row * 4 + o] + s;
      }
      __syncthreads();
    }
  }
}

extern "C" void kernel_launch(void* const* d_in, const int* in_sizes, int n_in,
                              void* d_out, int out_size, void* d_ws, size_t ws_size,
                              hipStream_t stream) {
  (void)in_sizes; (void)n_in; (void)out_size; (void)ws_size;
  const float* x        = (const float*)d_in[0];
  const float* enc_Wih  = (const float*)d_in[1];
  const float* enc_Whh  = (const float*)d_in[2];
  const float* enc_bih  = (const float*)d_in[3];
  const float* enc_bhh  = (const float*)d_in[4];
  const float* dec_Wih  = (const float*)d_in[5];
  const float* dec_Whh  = (const float*)d_in[6];
  const float* dec_bih  = (const float*)d_in[7];
  const float* dec_bhh  = (const float*)d_in[8];
  const float* fut_emb  = (const float*)d_in[9];
  const float* head_W   = (const float*)d_in[10];
  const float* head_b   = (const float*)d_in[11];

  float* out   = (float*)d_out;
  float* xnorm = out;
  float* pred  = out + (size_t)B_ * T_ * C_;

  // workspace layout
  float* feat = (float*)d_ws;                               // 16384*30*7 f32
  float* last = feat + (size_t)B_ * C_ * PH * NIN;          // 16384*4 f32
  unsigned short* encWbf = (unsigned short*)(last + (size_t)B_ * C_ * 4);
  unsigned short* decWbf = encWbf + (size_t)3 * HID * HID;  // 768*256 bf16 each

  const int nW = 3 * HID * HID;  // 196608
  conv_bf16_kernel<<<(nW + 255) / 256, 256, 0, stream>>>(enc_Whh, dec_Whh,
                                                         encWbf, decWbf, nW);

  swt_stats_kernel<<<B_ * (C_ / TC), 256, 0, stream>>>(x, xnorm, feat, last);

  gru_wmma_kernel<<<(B_ * C_) / MT, 256, 0, stream>>>(
      feat, last, encWbf, decWbf,
      enc_Wih, enc_bih, enc_bhh,
      dec_Wih, dec_bih, dec_bhh,
      fut_emb, head_W, head_b, pred);
}
